// PackedAttentionMoT_20109036880038
// MI455X (gfx1250) — compile-verified, matrix operand order checked by reading, not run on hardware
//
#include <hip/hip_runtime.h>

// ---------------------------------------------------------------------------
// CDNA5 / gfx1250 implementation of PackedAttentionMoT.
// All matmuls on v_wmma_f32_16x16x32_bf16 (wave32 WMMA), operands pre-converted
// to bf16 once to halve HBM/L2 traffic; fp32 accumulation throughout.
// GEMM staging uses GLOBAL_LOAD_ASYNC_TO_LDS_B128 (ASYNCcnt path) when the
// toolchain exposes the builtin, else falls back to vector load+ds_store.
// ---------------------------------------------------------------------------

typedef __attribute__((ext_vector_type(16))) __bf16 v16bf;
typedef __attribute__((ext_vector_type(8)))  __bf16 v8bf;
typedef __attribute__((ext_vector_type(4)))  __bf16 v4bf;
typedef __attribute__((ext_vector_type(8)))  float  v8f;
typedef __attribute__((ext_vector_type(4)))  int    v4i;

constexpr int kHID = 3584;
constexpr int kNH  = 28;
constexpr int kNKV = 4;
constexpr int kHD  = 128;
constexpr int kL   = 1024;
constexpr int kT   = 4096;

#define WMMA_BF16(A,B,C) \
  __builtin_amdgcn_wmma_f32_16x16x32_bf16(false,(A),false,(B),(short)0,(C),false,false)

#define AS1 __attribute__((address_space(1)))
#define AS3 __attribute__((address_space(3)))

// --- 16-byte global->LDS copy: async (ASYNCcnt) when available --------------
#if __has_builtin(__builtin_amdgcn_global_load_async_to_lds_b128)
#define HAVE_ASYNC_LDS 1
__device__ __forceinline__ void cp16_g2l(const void* g, void* l) {
  // builtin signature: (v4i AS1* src, v4i AS3* dst, imm offset, imm cpol)
  // flat LDS address low 32 bits are the LDS offset (ISA 10.2 aperture rules)
  __builtin_amdgcn_global_load_async_to_lds_b128(
      (AS1 v4i*)(unsigned long long)g,
      (AS3 v4i*)(unsigned int)(unsigned long long)l, 0, 0);
}
__device__ __forceinline__ void cp16_wait() {
#if __has_builtin(__builtin_amdgcn_s_wait_asynccnt)
  __builtin_amdgcn_s_wait_asynccnt(0);
#else
  asm volatile("s_wait_asynccnt 0" ::: "memory");
#endif
}
#else
#define HAVE_ASYNC_LDS 0
__device__ __forceinline__ void cp16_g2l(const void* g, void* l) {
  *(v8bf*)l = *(const v8bf*)g;
}
__device__ __forceinline__ void cp16_wait() {}
#endif

// Load one 16x32 (A-layout) or 32x16 (B-layout) bf16 fragment from a tile
// stored row-major with `pitch` elements per row.  Per ISA 7.12.2:
//   lane L (0-15):  row = L,      K elems {0..7, 16..23}
//   lane L (16-31): row = L-16,   K elems {8..15, 24..31}
__device__ __forceinline__ v16bf frag_ld(const __bf16* base, int pitch, int lane) {
  int row = lane & 15;
  int k0  = (lane & 16) ? 8 : 0;
  const __bf16* p = base + row * pitch + k0;
  v8bf lo = *(const v8bf*)(p);
  v8bf hi = *(const v8bf*)(p + 16);
  return __builtin_shufflevector(lo, hi, 0,1,2,3,4,5,6,7,8,9,10,11,12,13,14,15);
}

// ---------------------------------------------------------------------------
// f32 -> bf16 conversion (vectorized x4)
// ---------------------------------------------------------------------------
__global__ void cvt_f32_bf16(const float* __restrict__ in, __bf16* __restrict__ out,
                             long n) {
  long i = ((long)blockIdx.x * blockDim.x + threadIdx.x) * 4;
  if (i + 3 < n) {
    float4 f = *(const float4*)(in + i);
    v4bf o = { (__bf16)f.x, (__bf16)f.y, (__bf16)f.z, (__bf16)f.w };
    *(v4bf*)(out + i) = o;
  } else {
    for (long j = i; j < n; ++j) out[j] = (__bf16)in[j];
  }
}

// ---------------------------------------------------------------------------
// MoT GEMM:  Out[2*m+parity, n] = sum_k A[2*m+parity, k] * W[n, k] + bias[n]
// A: (kT, K) bf16, W: (N, K) bf16 row-major, Out: (kT, ldo) f32.
// Block tile 128(M) x 128(N) x 32(K); 8 waves, each wave a 32x64 tile
// (2 A-frags x 4 B-frags = 8 WMMAs per k-step).  grid = (N/128, 2048/128)
// ---------------------------------------------------------------------------
constexpr int GP = 40;  // padded LDS pitch (conflict-free fragment loads)

__global__ __launch_bounds__(256) void gemm_mot_bf16(
    const __bf16* __restrict__ A, const __bf16* __restrict__ W,
    const float* __restrict__ bias, float* __restrict__ Out,
    int K, int ldo, int parity) {
  __shared__ __align__(16) __bf16 As[128 * GP];
  __shared__ __align__(16) __bf16 Bs[128 * GP];

  int tid  = threadIdx.x;
  int lane = tid & 31;
  int w    = tid >> 5;
  int wm   = w >> 1;      // 0..3 : 32-row strip
  int wn   = w & 1;       // 0..1 : 64-col strip

  // global->LDS staging: each thread owns one row-half (16 elems = 2x16B)
  int lr = tid >> 1, lc = (tid & 1) * 16;
  long arow = ((long)(blockIdx.y * 128 + lr)) * 2 + parity;
  const __bf16* aptr = A + arow * (long)K + lc;
  const __bf16* wptr = W + ((long)(blockIdx.x * 128 + lr)) * K + lc;
  __bf16* asl = &As[lr * GP + lc];
  __bf16* bsl = &Bs[lr * GP + lc];

  v8f acc[8];
  for (int t = 0; t < 8; ++t)
    for (int i = 0; i < 8; ++i) acc[t][i] = 0.0f;

  for (int k0 = 0; k0 < K; k0 += 32) {
    cp16_g2l(aptr + k0,     asl);
    cp16_g2l(aptr + k0 + 8, asl + 8);
    cp16_g2l(wptr + k0,     bsl);
    cp16_g2l(wptr + k0 + 8, bsl + 8);
    __builtin_prefetch(aptr + k0 + 64, 0, 1);   // global_prefetch_b8
    __builtin_prefetch(wptr + k0 + 64, 0, 1);
    cp16_wait();
    __syncthreads();

    v16bf a0 = frag_ld(&As[(wm * 32)      * GP], GP, lane);
    v16bf a1 = frag_ld(&As[(wm * 32 + 16) * GP], GP, lane);
    v16bf b0 = frag_ld(&Bs[(wn * 64)      * GP], GP, lane);
    v16bf b1 = frag_ld(&Bs[(wn * 64 + 16) * GP], GP, lane);
    v16bf b2 = frag_ld(&Bs[(wn * 64 + 32) * GP], GP, lane);
    v16bf b3 = frag_ld(&Bs[(wn * 64 + 48) * GP], GP, lane);
    acc[0] = WMMA_BF16(a0, b0, acc[0]);
    acc[1] = WMMA_BF16(a0, b1, acc[1]);
    acc[2] = WMMA_BF16(a0, b2, acc[2]);
    acc[3] = WMMA_BF16(a0, b3, acc[3]);
    acc[4] = WMMA_BF16(a1, b0, acc[4]);
    acc[5] = WMMA_BF16(a1, b1, acc[5]);
    acc[6] = WMMA_BF16(a1, b2, acc[6]);
    acc[7] = WMMA_BF16(a1, b3, acc[7]);
    __syncthreads();
  }

  // Epilogue: C layout = lane(col), VGPR i (row i or i+8 by lane half)
  int coll = lane & 15, half = lane >> 4;
  for (int im = 0; im < 2; ++im)
    for (int jn = 0; jn < 4; ++jn) {
      v8f c = acc[im * 4 + jn];
      int col = blockIdx.x * 128 + wn * 64 + jn * 16 + coll;
      float bs = bias ? bias[col] : 0.0f;
      long rbase = (long)blockIdx.y * 128 + wm * 32 + im * 16 + half * 8;
      for (int i = 0; i < 8; ++i) {
        long row = (rbase + i) * 2 + parity;
        Out[row * (long)ldo + col] = c[i] + bs;
      }
    }
}

// ---------------------------------------------------------------------------
// Fused per-head RMSNorm + RoPE + scale, f32 in -> bf16 out.
// One wave per (token, head); lane handles 4 contiguous dims (lane^16 = d+-64).
// ---------------------------------------------------------------------------
__global__ __launch_bounds__(256) void norm_rope(
    const float* __restrict__ X, const float* __restrict__ wn0,
    const float* __restrict__ wn1, const float* __restrict__ cosb,
    const float* __restrict__ sinb, __bf16* __restrict__ Y, int nh, float scale) {
  int gw   = blockIdx.x * 8 + (threadIdx.x >> 5);
  int lane = threadIdx.x & 31;
  int tok = gw / nh, head = gw % nh;
  int d = lane * 4;

  const float* xp = X + ((long)tok * nh + head) * kHD + d;
  float4 x = *(const float4*)xp;
  float ss = x.x * x.x + x.y * x.y + x.z * x.z + x.w * x.w;
  for (int m = 1; m < 32; m <<= 1) ss += __shfl_xor(ss, m, 32);
  float inv = rsqrtf(ss * (1.0f / (float)kHD) + 1e-6f);

  const float* wn = (tok & 1) ? wn1 : wn0;
  float4 wv = *(const float4*)(wn + d);
  float y0 = x.x * wv.x * inv, y1 = x.y * wv.y * inv;
  float y2 = x.z * wv.z * inv, y3 = x.w * wv.w * inv;

  float sgn = (lane < 16) ? -1.0f : 1.0f;             // rotate_half sign
  float p0 = __shfl_xor(y0, 16, 32) * sgn;
  float p1 = __shfl_xor(y1, 16, 32) * sgn;
  float p2 = __shfl_xor(y2, 16, 32) * sgn;
  float p3 = __shfl_xor(y3, 16, 32) * sgn;

  float4 c = *(const float4*)(cosb + (long)tok * kHD + d);
  float4 s = *(const float4*)(sinb + (long)tok * kHD + d);
  v4bf o = { (__bf16)((y0 * c.x + p0 * s.x) * scale),
             (__bf16)((y1 * c.y + p1 * s.y) * scale),
             (__bf16)((y2 * c.z + p2 * s.z) * scale),
             (__bf16)((y3 * c.w + p3 * s.w) * scale) };
  *(v4bf*)(Y + ((long)tok * nh + head) * kHD + d) = o;
}

// ---------------------------------------------------------------------------
// Causal flash attention with GQA (7 q heads / kv head), bf16 WMMA, f32 accum.
// grid = (L/128, NH, S); block = 256 (8 waves, 16 q-rows each).
// kv tiles of 32 keys staged in LDS (K async); V staged transposed for the
// P@V B-fragment.
// ---------------------------------------------------------------------------
constexpr int KP = 136;  // Ks pitch (keys x d)
constexpr int VP = 40;   // Vt / Ps pitch

__global__ __launch_bounds__(256) void flash_attn_gqa(
    const __bf16* __restrict__ Q, const __bf16* __restrict__ Kb,
    const __bf16* __restrict__ Vb, __bf16* __restrict__ O) {
  __shared__ __align__(16) __bf16 Ks[32 * KP];        // [key][d]
  __shared__ __align__(16) __bf16 Vt[128 * VP];       // [d][key]
  __shared__ __align__(16) __bf16 Ps[8 * 16 * VP];    // per-wave P tile [row][key]

  int tid = threadIdx.x, lane = tid & 31, w = tid >> 5;
  int seq = blockIdx.z, head = blockIdx.y;
  int kvh = head / 7;
  int qbase = blockIdx.x * 128;          // first q row of block (within seq)
  int wq0   = qbase + w * 16;            // wave's first q row (within seq)
  long qtok0 = (long)seq * kL + wq0;

  // --- Q fragments (16 rows x 128 d = 4 k-chunks), loaded straight from global
  v16bf qf[4];
  {
    int row = lane & 15;
    int k0  = (lane & 16) ? 8 : 0;
    const __bf16* qp = Q + ((qtok0 + row) * kNH + head) * kHD;
    for (int f = 0; f < 4; ++f) {
      v8bf lo = *(const v8bf*)(qp + f * 32 + k0);
      v8bf hi = *(const v8bf*)(qp + f * 32 + k0 + 16);
      qf[f] = __builtin_shufflevector(lo, hi, 0,1,2,3,4,5,6,7,8,9,10,11,12,13,14,15);
    }
  }

  v8f acc[8];
  for (int t = 0; t < 8; ++t)
    for (int i = 0; i < 8; ++i) acc[t][i] = 0.0f;
  float m_i[8], l_i[8];
  for (int i = 0; i < 8; ++i) { m_i[i] = -__builtin_inff(); l_i[i] = 0.0f; }

  int coll = lane & 15, half = lane >> 4;
  int ntiles = (qbase + 128) >> 5;       // causal: keys up to block's last row

  for (int j = 0; j < ntiles; ++j) {
    int kv0 = j * 32;
    {  // stage K tile [32][128] (async path) and transposed V tile [128][32]
      int key = tid >> 3;
      int d0  = (tid & 7) * 16;
      long ktok = (long)seq * kL + kv0 + key;
      const __bf16* kp = Kb + (ktok * kNKV + kvh) * kHD + d0;
      __bf16* ksl = &Ks[key * KP + d0];
      cp16_g2l(kp, ksl);
      cp16_g2l(kp + 8, ksl + 8);
      const __bf16* vp = Vb + (ktok * kNKV + kvh) * kHD + d0;
      v8bf v0 = *(const v8bf*)(vp);
      v8bf v1 = *(const v8bf*)(vp + 8);
      for (int e = 0; e < 8; ++e) {
        Vt[(d0 + e) * VP + key]     = v0[e];
        Vt[(d0 + 8 + e) * VP + key] = v1[e];
      }
      cp16_wait();
    }
    __syncthreads();

    if (kv0 <= wq0 + 15) {  // wave-uniform guard (EXEC stays all-ones for WMMA)
      // scores S = Q @ K^T  (16 x 32, 2 C tiles, 4 k-chunks over d)
      v8f s0, s1;
      for (int i = 0; i < 8; ++i) { s0[i] = 0.0f; s1[i] = 0.0f; }
      for (int f = 0; f < 4; ++f) {
        v16bf b0 = frag_ld(&Ks[0 * KP + f * 32], KP, lane);
        v16bf b1 = frag_ld(&Ks[16 * KP + f * 32], KP, lane);
        s0 = WMMA_BF16(qf[f], b0, s0);
        s1 = WMMA_BF16(qf[f], b1, s1);
      }

      int key0 = kv0 + coll, key1 = kv0 + 16 + coll;
      float p0[8], p1[8], alpha[8];
      for (int i = 0; i < 8; ++i) {
        int qrow = wq0 + half * 8 + i;
        float v0 = (key0 <= qrow) ? s0[i] : -__builtin_inff();
        float v1 = (key1 <= qrow) ? s1[i] : -__builtin_inff();
        float mx = fmaxf(v0, v1);
        mx = fmaxf(mx, __shfl_xor(mx, 1, 32));
        mx = fmaxf(mx, __shfl_xor(mx, 2, 32));
        mx = fmaxf(mx, __shfl_xor(mx, 4, 32));
        mx = fmaxf(mx, __shfl_xor(mx, 8, 32));
        float mn = fmaxf(m_i[i], mx);
        float a, e0, e1;
        if (mn == -__builtin_inff()) { a = 1.0f; e0 = 0.0f; e1 = 0.0f; }
        else {
          a  = __expf(m_i[i] - mn);
          e0 = __expf(v0 - mn);
          e1 = __expf(v1 - mn);
        }
        float rs = e0 + e1;
        rs += __shfl_xor(rs, 1, 32);
        rs += __shfl_xor(rs, 2, 32);
        rs += __shfl_xor(rs, 4, 32);
        rs += __shfl_xor(rs, 8, 32);
        l_i[i] = l_i[i] * a + rs;
        m_i[i] = mn;
        alpha[i] = a;
        p0[i] = e0; p1[i] = e1;
      }
      for (int t = 0; t < 8; ++t)
        for (int i = 0; i < 8; ++i) acc[t][i] *= alpha[i];

      // C-layout -> A-layout reshape of P through per-wave LDS (in-order per wave)
      __bf16* pp = &Ps[w * 16 * VP];
      for (int i = 0; i < 8; ++i) {
        pp[(half * 8 + i) * VP + coll]      = (__bf16)p0[i];
        pp[(half * 8 + i) * VP + 16 + coll] = (__bf16)p1[i];
      }
      v16bf pa = frag_ld(pp, VP, lane);
      for (int t = 0; t < 8; ++t) {
        v16bf bv = frag_ld(&Vt[(t * 16) * VP], VP, lane);
        acc[t] = WMMA_BF16(pa, bv, acc[t]);
      }
    }
    __syncthreads();
  }

  // normalize and store (bf16, layout (T, NH*HD))
  float linv[8];
  for (int i = 0; i < 8; ++i) linv[i] = (l_i[i] > 0.0f) ? 1.0f / l_i[i] : 0.0f;
  for (int t = 0; t < 8; ++t)
    for (int i = 0; i < 8; ++i) {
      long tok = qtok0 + half * 8 + i;
      O[(tok * kNH + head) * kHD + t * 16 + coll] = (__bf16)(acc[t][i] * linv[i]);
    }
}

// ---------------------------------------------------------------------------
// Host-side orchestration
// ---------------------------------------------------------------------------
extern "C" void kernel_launch(void* const* d_in, const int* in_sizes, int n_in,
                              void* d_out, int out_size, void* d_ws, size_t ws_size,
                              hipStream_t stream) {
  const float* x    = (const float*)d_in[0];
  const float* cosb = (const float*)d_in[1];
  const float* sinb = (const float*)d_in[2];
  const float* wq   = (const float*)d_in[3];
  const float* bq   = (const float*)d_in[4];
  const float* wk   = (const float*)d_in[5];
  const float* bk   = (const float*)d_in[6];
  const float* wv   = (const float*)d_in[7];
  const float* bv   = (const float*)d_in[8];
  const float* wo   = (const float*)d_in[9];
  const float* wqg  = (const float*)d_in[10];
  const float* bqg  = (const float*)d_in[11];
  const float* wkg  = (const float*)d_in[12];
  const float* bkg  = (const float*)d_in[13];
  const float* wvg  = (const float*)d_in[14];
  const float* bvg  = (const float*)d_in[15];
  const float* wog  = (const float*)d_in[16];
  const float* qn   = (const float*)d_in[17];
  const float* kn   = (const float*)d_in[18];
  const float* qng  = (const float*)d_in[19];
  const float* kng  = (const float*)d_in[20];
  float* out = (float*)d_out;

  char* ws = (char*)d_ws;
  size_t off = 0;
  auto take = [&](size_t bytes) {
    char* p = ws + off;
    off += (bytes + 255) & ~(size_t)255;
    return p;
  };
  const long NQ = (long)kNH * kHD * kHID;    // q/o weight elems
  const long NK = (long)kNKV * kHD * kHID;   // k/v weight elems
  __bf16* xb    = (__bf16*)take((size_t)kT * kHID * 2);
  __bf16* wqb   = (__bf16*)take(NQ * 2);
  __bf16* wqgb  = (__bf16*)take(NQ * 2);
  __bf16* wkb   = (__bf16*)take(NK * 2);
  __bf16* wkgb  = (__bf16*)take(NK * 2);
  __bf16* wvb   = (__bf16*)take(NK * 2);
  __bf16* wvgb  = (__bf16*)take(NK * 2);
  __bf16* wob   = (__bf16*)take(NQ * 2);
  __bf16* wogb  = (__bf16*)take(NQ * 2);
  float*  qf    = (float*)take((size_t)kT * kNH * kHD * 4);
  float*  kf    = (float*)take((size_t)kT * kNKV * kHD * 4);
  float*  vf    = (float*)take((size_t)kT * kNKV * kHD * 4);
  __bf16* qb    = (__bf16*)take((size_t)kT * kNH * kHD * 2);
  __bf16* kb    = (__bf16*)take((size_t)kT * kNKV * kHD * 2);
  __bf16* vb    = (__bf16*)take((size_t)kT * kNKV * kHD * 2);
  __bf16* attnb = (__bf16*)take((size_t)kT * kNH * kHD * 2);

  auto cvt = [&](const float* src, __bf16* dst, long n) {
    unsigned blocks = (unsigned)((n + 1023) / 1024);
    cvt_f32_bf16<<<blocks, 256, 0, stream>>>(src, dst, n);
  };

  // 1) one-time bf16 conversions (weights + activations)
  cvt(x,   xb,   (long)kT * kHID);
  cvt(wq,  wqb,  NQ);  cvt(wqg, wqgb, NQ);
  cvt(wk,  wkb,  NK);  cvt(wkg, wkgb, NK);
  cvt(wv,  wvb,  NK);  cvt(wvg, wvgb, NK);
  cvt(wo,  wob,  NQ);  cvt(wog, wogb, NQ);

  // 2) QKV projections, even (parity 0 = und) / odd (parity 1 = gen)
  dim3 gq(kNH * kHD / 128, kT / 2 / 128);   // (28,16)
  dim3 gk(kNKV * kHD / 128, kT / 2 / 128);  // (4,16)
  gemm_mot_bf16<<<gq, 256, 0, stream>>>(xb, wqb,  bq,  qf, kHID, kNH * kHD, 0);
  gemm_mot_bf16<<<gq, 256, 0, stream>>>(xb, wqgb, bqg, qf, kHID, kNH * kHD, 1);
  gemm_mot_bf16<<<gk, 256, 0, stream>>>(xb, wkb,  bk,  kf, kHID, kNKV * kHD, 0);
  gemm_mot_bf16<<<gk, 256, 0, stream>>>(xb, wkgb, bkg, kf, kHID, kNKV * kHD, 1);
  gemm_mot_bf16<<<gk, 256, 0, stream>>>(xb, wvb,  bv,  vf, kHID, kNKV * kHD, 0);
  gemm_mot_bf16<<<gk, 256, 0, stream>>>(xb, wvgb, bvg, vf, kHID, kNKV * kHD, 1);

  // 3) RMSNorm + RoPE (+1/sqrt(HD) folded into q), bf16 out; plain cvt for v
  norm_rope<<<kT * kNH / 8, 256, 0, stream>>>(qf, qn, qng, cosb, sinb, qb, kNH,
                                              0.08838834764831845f);
  norm_rope<<<kT * kNKV / 8, 256, 0, stream>>>(kf, kn, kng, cosb, sinb, kb, kNKV,
                                               1.0f);
  cvt(vf, vb, (long)kT * kNKV * kHD);

  // 4) causal flash attention with GQA
  flash_attn_gqa<<<dim3(kL / 128, kNH, 4), 256, 0, stream>>>(qb, kb, vb, attnb);

  // 5) output projection per parity -> fp32 d_out
  gemm_mot_bf16<<<dim3(kHID / 128, kT / 2 / 128), 256, 0, stream>>>(
      attnb, wob, nullptr, out, kNH * kHD, kHID, 0);
  gemm_mot_bf16<<<dim3(kHID / 128, kT / 2 / 128), 256, 0, stream>>>(
      attnb, wogb, nullptr, out, kNH * kHD, kHID, 1);
}
